// Model_39676907885804
// MI455X (gfx1250) — matrix operand order repaired and checked by reading, not compiled, not run on hardware
//
#include <hip/hip_runtime.h>

typedef __bf16 bf16;
typedef __attribute__((ext_vector_type(16))) __bf16 v16bf;
typedef __attribute__((ext_vector_type(8)))  float  v8f;

#define BM 128
#define BN 128
#define BK 64
#define LDSROW 72   // BK + 8 pad (144B stride, multiple of 16B)

union FragAB { uint4 u[2]; v16bf v; };
union Pack32 { uint4 u[4]; bf16 h[32]; };
union Pack16 { uint4 u[2]; bf16 h[16]; };

// 64B contiguous global -> LDS async copy (4 x B128), tracked by ASYNCcnt.
__device__ __forceinline__ void async_copy64B(unsigned lds_off, const void* gptr) {
  asm volatile(
      "global_load_async_to_lds_b128 %0, %1, off\n\t"
      "global_load_async_to_lds_b128 %0, %1, off offset:16\n\t"
      "global_load_async_to_lds_b128 %0, %1, off offset:32\n\t"
      "global_load_async_to_lds_b128 %0, %1, off offset:48"
      :: "v"(lds_off), "v"(gptr) : "memory");
}
__device__ __forceinline__ void wait_asynccnt0() {
  asm volatile("s_wait_asynccnt 0" ::: "memory");
}

__global__ __launch_bounds__(256) void cvt_bf16_kernel(
    const float* __restrict__ in, bf16* __restrict__ out, int n) {
  int i = blockIdx.x * 256 + threadIdx.x;
  if (i < n) out[i] = (bf16)in[i];
}

// out[c][r] = in[r][c]  (store transposed bf16 copy)
__global__ __launch_bounds__(256) void cvt_bf16_T_kernel(
    const float* __restrict__ in, bf16* __restrict__ out, int rows, int cols) {
  int i = blockIdx.x * 256 + threadIdx.x;
  if (i < rows * cols) {
    int r = i / cols, c = i - r * cols;
    out[(size_t)c * rows + r] = (bf16)in[i];
  }
}

// Batched GEMM  C[b] = (A[b] @ B[b]) * scalarScale * colscale[n]
//  A: M x K row-major (f32 or bf16); B either as Bt (N x K row-major,
//  b_row_major==0, async-DMAed to LDS) or as B (K x N row-major,
//  b_row_major==1, transposed through LDS).  C: M x N (f32 or bf16).
//  M%128==0, N%128==0, K%64==0.  Double-buffered LDS pipeline.
__global__ __launch_bounds__(256) void gemm_bf16_wmma(
    const void* __restrict__ Aptr, int a_is_f32, long long strideA,
    const bf16* __restrict__ Bptr, int b_row_major, long long strideB,
    void* __restrict__ Cptr, int c_is_f32, long long strideC,
    int M, int N, int K,
    const float* __restrict__ colscale, float scalarScale)
{
  __shared__ bf16 As[2][BM][LDSROW];
  __shared__ bf16 Bs[2][BN][LDSROW];

  const int t    = threadIdx.x;
  const int lane = t & 31;
  const int wid  = t >> 5;
  const int wm   = wid & 3;    // 4 wave-rows of 32
  const int wn   = wid >> 2;   // 2 wave-cols of 64
  const int lrow = lane & 15;
  const int lhi  = lane >> 4;

  const int tileN0 = blockIdx.x * BN;
  const int tileM0 = blockIdx.y * BM;
  const size_t offA = (size_t)blockIdx.z * (size_t)strideA;
  const size_t offB = (size_t)blockIdx.z * (size_t)strideB;
  const size_t offC = (size_t)blockIdx.z * (size_t)strideC;

  v8f acc[2][4];
  v8f zero;
  #pragma unroll
  for (int e = 0; e < 8; e++) zero[e] = 0.0f;
  #pragma unroll
  for (int i = 0; i < 2; i++)
    #pragma unroll
    for (int j = 0; j < 4; j++) acc[i][j] = zero;

  const int ar = t >> 1;         // 0..127 staging row
  const int ac = (t & 1) * 32;   // staging k-offset (elements)
  const int bk = t >> 2;         // 0..63  (mode-N staging)
  const int bn = (t & 3) * 32;

  float4 areg[8];                // f32-A prefetch registers
  Pack32 breg;                   // mode-N B prefetch registers

  // Issue global traffic for buffer `buf`, K-offset k0 (async DMA or regs).
  auto stage_issue = [&](int buf, int k0) {
    if (a_is_f32) {
      const float* Af = (const float*)Aptr + offA + (size_t)(tileM0 + ar) * K + k0 + ac;
      #pragma unroll
      for (int i = 0; i < 8; i++) areg[i] = *(const float4*)(Af + 4 * i);
    } else {
      const bf16* Ah = (const bf16*)Aptr + offA + (size_t)(tileM0 + ar) * K + k0 + ac;
      async_copy64B((unsigned)(uintptr_t)&As[buf][ar][ac], Ah);
    }
    if (b_row_major) {
      const bf16* Bp = Bptr + offB + (size_t)(k0 + bk) * N + tileN0 + bn;
      #pragma unroll
      for (int i = 0; i < 4; i++) breg.u[i] = *(const uint4*)(Bp + 8 * i);
    } else {
      const bf16* Bh = Bptr + offB + (size_t)(tileN0 + ar) * K + k0 + ac;
      async_copy64B((unsigned)(uintptr_t)&Bs[buf][ar][ac], Bh);
    }
  };

  // Write register-staged data into LDS buffer `buf` (manual paths only).
  auto stage_commit = [&](int buf) {
    if (a_is_f32) {
      #pragma unroll
      for (int i = 0; i < 8; i++) {
        As[buf][ar][ac + 4 * i + 0] = (bf16)areg[i].x;
        As[buf][ar][ac + 4 * i + 1] = (bf16)areg[i].y;
        As[buf][ar][ac + 4 * i + 2] = (bf16)areg[i].z;
        As[buf][ar][ac + 4 * i + 3] = (bf16)areg[i].w;
      }
    }
    if (b_row_major) {
      #pragma unroll
      for (int i = 0; i < 32; i++) Bs[buf][bn + i][bk] = breg.h[i];
    }
  };

  auto compute_tile = [&](int buf) {
    #pragma unroll
    for (int ks = 0; ks < BK; ks += 32) {
      FragAB a[2], b[4];
      #pragma unroll
      for (int i = 0; i < 2; i++) {
        // A 16x32: lane<16 -> K{0..7,16..23}, lane>=16 -> K{8..15,24..31}
        const bf16* p = &As[buf][wm * 32 + i * 16 + lrow][ks + lhi * 8];
        a[i].u[0] = *(const uint4*)p;
        a[i].u[1] = *(const uint4*)(p + 16);
      }
      #pragma unroll
      for (int j = 0; j < 4; j++) {
        // B 32x16 fed as Bt rows: lane<16 -> K{0..7,16..23}, lane>=16 -> K{8..15,24..31}
        const bf16* p = &Bs[buf][wn * 64 + j * 16 + lrow][ks + lhi * 8];
        b[j].u[0] = *(const uint4*)p;
        b[j].u[1] = *(const uint4*)(p + 16);
      }
      #pragma unroll
      for (int i = 0; i < 2; i++)
        #pragma unroll
        for (int j = 0; j < 4; j++)
          acc[i][j] = __builtin_amdgcn_wmma_f32_16x16x32_bf16(
              false, a[i].v, false, b[j].v, (short)0, acc[i][j], false, false);
    }
  };

  // ---- double-buffered pipeline ----
  stage_issue(0, 0);
  stage_commit(0);
  wait_asynccnt0();
  __syncthreads();

  int cur = 0;
  for (int k0 = 0; k0 < K; k0 += BK) {
    const int nk = k0 + BK;
    if (nk < K) stage_issue(cur ^ 1, nk);   // overlap DMA with WMMA below
    compute_tile(cur);
    if (nk < K) {
      stage_commit(cur ^ 1);
      wait_asynccnt0();
    }
    __syncthreads();
    cur ^= 1;
  }

  // ---- epilogue: C/D layout VGPR v -> (M=v | v+8 split on lane halves) ----
  #pragma unroll
  for (int j = 0; j < 4; j++) {
    const int n = tileN0 + wn * 64 + j * 16 + lrow;
    float cs = scalarScale;
    if (colscale) cs *= colscale[n];
    #pragma unroll
    for (int i = 0; i < 2; i++) {
      const int m0 = tileM0 + wm * 32 + i * 16 + lhi * 8;
      #pragma unroll
      for (int r = 0; r < 8; r++) {
        const float val = acc[i][j][r] * cs;
        const size_t idx = offC + (size_t)(m0 + r) * N + n;
        if (c_is_f32) ((float*)Cptr)[idx] = val;
        else          ((bf16*)Cptr)[idx]  = (bf16)val;
      }
    }
  }
}

// Row softmax over 512 cols + deterministic dropout (keep-mask * 2.0).
// One wave32 per row, 16 elements per lane.
__global__ __launch_bounds__(256) void softmax_dropout_kernel(
    const bf16* __restrict__ x, const int* __restrict__ mask,
    bf16* __restrict__ y, int nrows)
{
  const int lane = threadIdx.x & 31;
  const int wid  = threadIdx.x >> 5;
  const int row  = blockIdx.x * 8 + wid;
  if (row >= nrows) return;
  const size_t base = (size_t)row * 512 + lane * 16;

  Pack16 d;
  d.u[0] = *(const uint4*)(x + base);
  d.u[1] = *(const uint4*)(x + base + 8);
  float v[16];
  float mx = -3.0e38f;
  #pragma unroll
  for (int i = 0; i < 16; i++) { v[i] = (float)d.h[i]; mx = fmaxf(mx, v[i]); }
  #pragma unroll
  for (int o = 16; o > 0; o >>= 1) mx = fmaxf(mx, __shfl_xor(mx, o, 32));
  float s = 0.0f;
  #pragma unroll
  for (int i = 0; i < 16; i++) { v[i] = __expf(v[i] - mx); s += v[i]; }
  #pragma unroll
  for (int o = 16; o > 0; o >>= 1) s += __shfl_xor(s, o, 32);
  const float r = 2.0f / s;   // 1/(1-DROPOUT_P) = 2
  Pack16 o2;
  #pragma unroll
  for (int i = 0; i < 16; i++)
    o2.h[i] = (bf16)(v[i] * r * (float)mask[base + i]);
  *(uint4*)(y + base)     = o2.u[0];
  *(uint4*)(y + base + 8) = o2.u[1];
}

extern "C" void kernel_launch(void* const* d_in, const int* in_sizes, int n_in,
                              void* d_out, int out_size, void* d_ws, size_t ws_size,
                              hipStream_t stream)
{
  (void)in_sizes; (void)n_in; (void)out_size; (void)ws_size;
  const int B = 8, H = 16, L = 512, D = 512;
  const float inv = 1.0f / (float)D;
  const size_t NE = (size_t)B * H * L * D;  // 33,554,432
  const size_t DD = (size_t)D * D;

  const float* query = (const float*)d_in[0];
  const float* key_in = (const float*)d_in[1];
  const float* value = (const float*)d_in[2];
  const float* Wqkv  = (const float*)d_in[3];  // 3 x D x D
  const float* posw  = (const float*)d_in[4];  // 1 x D x D
  const float* posb  = (const float*)d_in[5];  // 1 x D
  const float* pwd   = (const float*)d_in[6];  // D x D
  const float* pbd   = (const float*)d_in[7];  // D
  const int*   dmask = (const int*)d_in[8];    // B,H,L,L int32

  bf16* ws    = (bf16*)d_ws;
  bf16* qb    = ws;            // NE
  bf16* kb    = qb + NE;       // NE
  bf16* vb    = kb + NE;       // NE
  bf16* qkb   = vb + NE;       // NE (qk, later reused for softmax output s)
  bf16* Wb    = qkb + NE;      // 3*DD  (bf16 of in_proj rows)
  bf16* pwdT  = Wb + 3 * DD;   // DD
  bf16* poswT = pwdT + DD;     // DD
  bf16* WeffT = poswT + DD;    // 3*DD
  bf16* qk2b  = qb;            // alias: q is dead once qk is computed

  dim3 blk(256);

  // bf16 weight prep
  cvt_bf16_kernel<<<dim3((unsigned)((3 * DD + 255) / 256)), blk, 0, stream>>>(
      Wqkv, Wb, (int)(3 * DD));
  cvt_bf16_T_kernel<<<dim3((unsigned)((DD + 255) / 256)), blk, 0, stream>>>(
      pwd, pwdT, D, D);
  cvt_bf16_T_kernel<<<dim3((unsigned)((DD + 255) / 256)), blk, 0, stream>>>(
      posw, poswT, D, D);

  // Weff_i^T = pwd^T @ W_i^T   (Bt tile sourced directly from W_i rows)
  gemm_bf16_wmma<<<dim3(D / BN, D / BM, 3), blk, 0, stream>>>(
      pwdT, 0, 0LL, Wb, 0, (long long)DD,
      WeffT, 0, (long long)DD, D, D, D, nullptr, 1.0f);

  // Fused projections: x' = (x @ Weff) * inv^3 * pbd[n]
  const float inv3 = inv * inv * inv;
  const float* xin[3]  = {query, key_in, value};
  bf16*        xout[3] = {qb, kb, vb};
  for (int i = 0; i < 3; i++) {
    gemm_bf16_wmma<<<dim3(D / BN, (B * H * L) / BM, 1), blk, 0, stream>>>(
        xin[i], 1, 0LL, WeffT + (size_t)i * DD, 0, 0LL,
        xout[i], 0, 0LL, B * H * L, D, D, pbd, inv3);
  }

  // qk = q @ k^T per head (Bt = k rows directly)
  gemm_bf16_wmma<<<dim3(L / BN, L / BM, B * H), blk, 0, stream>>>(
      qb, 0, (long long)(L * D), kb, 0, (long long)(L * D),
      qkb, 0, (long long)(L * L), L, L, D, nullptr, 1.0f);

  // qk2 = (qk @ posw) * inv^2 * posb[n]
  gemm_bf16_wmma<<<dim3(L / BN, L / BM, B * H), blk, 0, stream>>>(
      qkb, 0, (long long)(L * L), poswT, 0, 0LL,
      qk2b, 0, (long long)(L * L), L, L, L, posb, inv * inv);

  // softmax + dropout -> s (overwrites qk buffer)
  softmax_dropout_kernel<<<dim3((B * H * L) / 8), blk, 0, stream>>>(
      qk2b, dmask, qkb, B * H * L);

  // out = s @ v  (v is K x N row-major -> transpose through LDS)
  gemm_bf16_wmma<<<dim3(D / BN, L / BM, B * H), blk, 0, stream>>>(
      qkb, 0, (long long)(L * L), vb, 1, (long long)(L * D),
      (float*)d_out, 1, (long long)(L * D), L, D, L, nullptr, 1.0f);
}
